// selfattention_43087111913617
// MI455X (gfx1250) — matrix operand order repaired and checked by reading, not compiled, not run on hardware
//
#include <hip/hip_runtime.h>

// ---------------------------------------------------------------------------
// Cross-attention (shared softmax) for MI455X / gfx1250, wave32 + WMMA.
//   Phase 1: 1x1-conv projections -> f16 Q[n][32], K[n][32], V1/V2 [c][n]
//   Phase 2: flash attention, S^T = K*Q^T via v_wmma_f32_16x16x32_f16,
//            online softmax in registers, P*V via the same WMMA shape,
//            gamma*O + residual written for BOTH outputs in one pass.
// Workspace: 2+2+16+16 = 36 MB of d_ws.
// ---------------------------------------------------------------------------

typedef __attribute__((ext_vector_type(16))) _Float16 v16h;
typedef __attribute__((ext_vector_type(8)))  _Float16 v8h;
typedef __attribute__((ext_vector_type(8)))  float    v8f;

constexpr int Bsz  = 8;
constexpr int Cch  = 256;
constexpr int Npix = 4096;   // 64*64
constexpr int Dqk  = 32;

#define DEVI static __device__ __forceinline__

// Two 16B (global_load_b128) chunks -> one 8-VGPR f16 WMMA operand.
DEVI v16h load_pair_b128(const _Float16* p, int second_off_halves) {
  union { v16h v; v8h h[2]; } u;
  u.h[0] = *(const v8h*)(p);
  u.h[1] = *(const v8h*)(p + second_off_halves);
  return u.v;
}

DEVI v8f wmma_f16(v16h a, v16h b, v8f c) {
  // D = A(16x32 f16) * B(32x16 f16) + C(16x16 f32)
  return __builtin_amdgcn_wmma_f32_16x16x32_f16(
      /*neg_a=*/false, a, /*neg_b=*/false, b,
      /*c_mod=*/(short)0, c, /*reuse_a=*/false, /*reuse_b=*/false);
}

// ---------------------------------------------------------------------------
// Phase 1: projections. grid = (Npix/64, 18, B), block = 64 (2 waves).
//   groups 0..7  : v1 channels 32g..32g+31  (input1 x Wv)  -> V1[c][n]
//   groups 8..15 : v2 channels               (input2 x Wv)  -> V2[c][n]
//   group  16    : q2 (input2 x Wq)                         -> Qh[n][32]
//   group  17    : k1 (input1 x Wk)                         -> Kh[n][32]
// W indices are thread-uniform -> scalar loads; X loads coalesced over n.
// ---------------------------------------------------------------------------
__global__ void __launch_bounds__(64) proj_kernel(
    const float* __restrict__ in1, const float* __restrict__ in2,
    const float* __restrict__ Wq, const float* __restrict__ bq,
    const float* __restrict__ Wk, const float* __restrict__ bk,
    const float* __restrict__ Wv, const float* __restrict__ bv,
    _Float16* __restrict__ Qh, _Float16* __restrict__ Kh,
    _Float16* __restrict__ V1, _Float16* __restrict__ V2) {
  const int n = blockIdx.x * 64 + threadIdx.x;
  const int g = blockIdx.y;
  const int b = blockIdx.z;

  const float* X; const float* W; const float* bias; int o0;
  if (g < 8)        { X = in1; W = Wv; bias = bv; o0 = g * 32; }
  else if (g < 16)  { X = in2; W = Wv; bias = bv; o0 = (g - 8) * 32; }
  else if (g == 16) { X = in2; W = Wq; bias = bq; o0 = 0; }
  else              { X = in1; W = Wk; bias = bk; o0 = 0; }

  float acc[32];
#pragma unroll
  for (int o = 0; o < 32; ++o) acc[o] = 0.f;

  const float* xp = X + (size_t)b * Cch * Npix + n;
  for (int c = 0; c < Cch; ++c) {
    const float x = xp[(size_t)c * Npix];
#pragma unroll
    for (int o = 0; o < 32; ++o)
      acc[o] = fmaf(W[(o0 + o) * Cch + c], x, acc[o]);
  }

  if (g < 16) {
    _Float16* V = (g < 8) ? V1 : V2;
#pragma unroll
    for (int o = 0; o < 32; ++o)
      V[(size_t)(b * Cch + o0 + o) * Npix + n] = (_Float16)(acc[o] + bias[o0 + o]);
  } else {
    _Float16* row = ((g == 16) ? Qh : Kh) + ((size_t)b * Npix + n) * Dqk;
#pragma unroll
    for (int o = 0; o < 32; ++o)
      row[o] = (_Float16)(acc[o] + bias[o]);
  }
}

// ---------------------------------------------------------------------------
// Phase 2: flash attention. grid = (Npix/16, B), block = 256 (8 waves).
// Wave w: 16 query rows m0..m0+15, 64 channels of (w<4 ? V1/out1 : V2/out2).
// Per 32-key step:
//   S^T tiles via WMMA (A = K rows, B = Q^T): lane = query col m, VGPR = key
//   row -> softmax is per-lane + one shfl_xor(16); exp'd S^T repacks IN-LANE
//   into the P A-operand; 4 P*V WMMAs accumulate the 16x64 O tile.
// ---------------------------------------------------------------------------
__global__ void __launch_bounds__(256) attn_kernel(
    const _Float16* __restrict__ Qh, const _Float16* __restrict__ Kh,
    const _Float16* __restrict__ V1, const _Float16* __restrict__ V2,
    const float* __restrict__ in1, const float* __restrict__ in2,
    const float* __restrict__ gamma, float* __restrict__ out) {
  const int b    = blockIdx.y;
  const int m0   = blockIdx.x * 16;
  const int wave = threadIdx.x >> 5;
  const int lane = threadIdx.x & 31;
  const int lm   = lane & 15;      // column index within 16x16 tiles
  const int hi   = lane >> 4;      // lane half

  const bool first = (wave < 4);
  const _Float16* V   = first ? V1 : V2;
  const float*    xin = first ? in1 : in2;
  float* outp = out + (first ? (size_t)0 : (size_t)Bsz * Cch * Npix);
  const int c0 = (wave & 3) * 64;

  // Q tile (rows m0..m0+15) as B operand of S^T = K*Q^T:
  // lane holds column m=lm; d = (0..7)+8*hi then (16..23)+8*hi.
  const _Float16* qrow = Qh + ((size_t)b * Npix + m0 + lm) * Dqk + hi * 8;
  const v16h qb = load_pair_b128(qrow, 16);

  v8f acc[4] = {};                  // O tile: 16 rows x 64 channels, f32
  float mrun = -__builtin_inff();   // running row max (row = lm)
  float srun = 0.f;                 // running row sum

  const _Float16* kbase = Kh + (size_t)b * Npix * Dqk;
  const _Float16* vbase = V  + (size_t)b * Cch * Npix;

  for (int n0 = 0; n0 < Npix; n0 += 32) {
    // K rows as A operands: lane holds row n0+lm; A layout interleaves
    // d = (0..7)+8*hi then (16..23)+8*hi -> two b128 at +0 / +32B.
    const _Float16* krow = kbase + (size_t)(n0 + lm) * Dqk + hi * 8;
    const v16h ka = load_pair_b128(krow, 16);                // keys n0..n0+15
    const v16h kb = load_pair_b128(krow + 16 * Dqk, 16);     // keys n0+16..31
    // V chunk (32 keys x 16 chans per tile) as B operands: lane holds channel
    // c0+t*16+lm; k(=n) = (0..7)+8*hi then (16..23)+8*hi in [c][n].
    const _Float16* vrow = vbase + (size_t)(c0 + lm) * Npix + n0 + hi * 8;
    const v16h vb0 = load_pair_b128(vrow, 16);
    const v16h vb1 = load_pair_b128(vrow + 16 * (size_t)Npix, 16);
    const v16h vb2 = load_pair_b128(vrow + 32 * (size_t)Npix, 16);
    const v16h vb3 = load_pair_b128(vrow + 48 * (size_t)Npix, 16);
    if (n0 + 32 < Npix) __builtin_prefetch(krow + 32 * Dqk, 0, 1);

    v8f z{};
    v8f sta = wmma_f16(ka, qb, z);  // S^T: keys n0..n0+15   x queries
    v8f stb = wmma_f16(kb, qb, z);  // S^T: keys n0+16..n0+31 x queries

    // Online softmax for query row m = lm (each lane owns one column).
    float mloc = sta[0];
#pragma unroll
    for (int i = 1; i < 8; ++i) mloc = fmaxf(mloc, sta[i]);
#pragma unroll
    for (int i = 0; i < 8; ++i) mloc = fmaxf(mloc, stb[i]);
    mloc = fmaxf(mloc, __shfl_xor(mloc, 16));
    const float mnew = fmaxf(mrun, mloc);
    const float corr = __expf(mrun - mnew);

    // exp + IN-LANE repack into P's f16 A-operand:
    //   A vgpr v<4  <- (sta[2v],  sta[2v+1])   (k = (0..7)+8*hi)
    //   A vgpr v>=4 <- (stb[2v'], stb[2v'+1])  (k = (16..23)+8*hi)
    v16h pA;
    float ssum = 0.f;
#pragma unroll
    for (int i = 0; i < 8; ++i) {
      const float ea = __expf(sta[i] - mnew);
      const float eb = __expf(stb[i] - mnew);
      ssum += ea + eb;
      pA[i]     = (_Float16)ea;
      pA[8 + i] = (_Float16)eb;
    }
    ssum += __shfl_xor(ssum, 16);
    srun = srun * corr + ssum;
    mrun = mnew;

    // Rescale accumulators: O-tile row for vgpr v is m = v + 8*hi, whose
    // corr lives in lane (v + 8*hi) (rows are replicated across lane halves).
#pragma unroll
    for (int v = 0; v < 8; ++v) {
      const float cr = __shfl(corr, v + 8 * hi);
#pragma unroll
      for (int t = 0; t < 4; ++t) acc[t][v] *= cr;
    }

    acc[0] = wmma_f16(pA, vb0, acc[0]);
    acc[1] = wmma_f16(pA, vb1, acc[1]);
    acc[2] = wmma_f16(pA, vb2, acc[2]);
    acc[3] = wmma_f16(pA, vb3, acc[3]);
  }

  // Normalize, apply gamma, residual-add, store (8 consecutive m per lane).
  float rinv[8];
#pragma unroll
  for (int v = 0; v < 8; ++v)
    rinv[v] = 1.0f / __shfl(srun, v + 8 * hi);
  const float ga = gamma[0];

#pragma unroll
  for (int t = 0; t < 4; ++t) {
    const int c = c0 + t * 16 + lm;
    const size_t rowoff = (size_t)(b * Cch + c) * Npix + m0 + 8 * hi;
    const float* xr = xin + rowoff;
    float* orow = outp + rowoff;
#pragma unroll
    for (int v = 0; v < 8; ++v)
      orow[v] = fmaf(ga, acc[t][v] * rinv[v], xr[v]);
  }
}

// ---------------------------------------------------------------------------
extern "C" void kernel_launch(void* const* d_in, const int* in_sizes, int n_in,
                              void* d_out, int out_size, void* d_ws, size_t ws_size,
                              hipStream_t stream) {
  (void)in_sizes; (void)n_in; (void)out_size; (void)ws_size;
  const float* in1 = (const float*)d_in[0];
  const float* in2 = (const float*)d_in[1];
  const float* Wq  = (const float*)d_in[2];
  const float* bq  = (const float*)d_in[3];
  const float* Wk  = (const float*)d_in[4];
  const float* bk  = (const float*)d_in[5];
  const float* Wv  = (const float*)d_in[6];
  const float* bv  = (const float*)d_in[7];
  const float* gm  = (const float*)d_in[8];

  char* ws = (char*)d_ws;
  const size_t qkBytes = (size_t)Bsz * Npix * Dqk * sizeof(_Float16); //  2 MB
  const size_t vBytes  = (size_t)Bsz * Cch * Npix * sizeof(_Float16); // 16 MB
  _Float16* Qh = (_Float16*)(ws);
  _Float16* Kh = (_Float16*)(ws + qkBytes);
  _Float16* V1 = (_Float16*)(ws + 2 * qkBytes);
  _Float16* V2 = (_Float16*)(ws + 2 * qkBytes + vBytes);

  proj_kernel<<<dim3(Npix / 64, 18, Bsz), 64, 0, stream>>>(
      in1, in2, Wq, bq, Wk, bk, Wv, bv, Qh, Kh, V1, V2);
  attn_kernel<<<dim3(Npix / 16, Bsz), 256, 0, stream>>>(
      Qh, Kh, V1, V2, in1, in2, gm, (float*)d_out);
}
